// MultiHeadedSelfAttention_3839700762866
// MI455X (gfx1250) — matrix operand order repaired and checked by reading, not compiled, not run on hardware
//
#include <hip/hip_runtime.h>
#include <hip/hip_bf16.h>

typedef __attribute__((ext_vector_type(16))) _Float16 v16h;
typedef __attribute__((ext_vector_type(8)))  _Float16 v8h;
typedef __attribute__((ext_vector_type(8)))  float    v8f;

#define NB  2
#define SEQ 2048
#define EMB 1024
#define NH  16
#define HD  64

static __device__ inline v8f v8f_zero() {
  v8f z = {0.f, 0.f, 0.f, 0.f, 0.f, 0.f, 0.f, 0.f};
  return z;
}

// ---------------------------------------------------------------------------
// Kernel 1: per-head projections  qq = k@Wq^T, kk = k@Wk^T, vvT = (k@Wv^T)^T
// f32 math, f16 outputs. vvT stored [n][h][d][s] so P@V B-operand loads are
// contiguous. qq/kk stored [n][h][s][d].
// ---------------------------------------------------------------------------
__global__ void proj_kernel(const float* __restrict__ key,
                            const float* __restrict__ Wk,
                            const float* __restrict__ Wq,
                            const float* __restrict__ Wv,
                            _Float16* __restrict__ qq,
                            _Float16* __restrict__ kk,
                            _Float16* __restrict__ vvT) {
  int g    = blockIdx.x * blockDim.x + threadIdx.x;
  int dout = g & 63;
  int h    = (g >> 6) & 15;
  int s    = (g >> 10) & 2047;
  int n    = g >> 21;

  const float* x  = key + ((size_t)(n * SEQ + s)) * EMB + h * HD;
  const float* wq = Wq + dout * HD;
  const float* wk = Wk + dout * HD;
  const float* wv = Wv + dout * HD;

  float aq = 0.f, ak = 0.f, av = 0.f;
#pragma unroll 8
  for (int d = 0; d < HD; ++d) {
    float xv = x[d];
    aq += xv * wq[d];
    ak += xv * wk[d];
    av += xv * wv[d];
  }
  size_t hb = ((size_t)(n * NH + h)) * SEQ;
  qq[(hb + s) * HD + dout] = (_Float16)aq;
  kk[(hb + s) * HD + dout] = (_Float16)ak;
  vvT[(((size_t)(n * NH + h)) * HD + dout) * SEQ + s] = (_Float16)av;
}

// ---------------------------------------------------------------------------
// Kernel 2: Wo f32 -> f16 (layout preserved, row-major [e][e'])
// ---------------------------------------------------------------------------
__global__ void cvt_wo(const float* __restrict__ src, _Float16* __restrict__ dst) {
  int i = blockIdx.x * blockDim.x + threadIdx.x;
  dst[i] = (_Float16)src[i];
}

// ---------------------------------------------------------------------------
// Kernel 3: attention core. One wave32 per (n, h, 16-row q tile).
//   energy tile = Q(16xd) @ K^T(dx16)  via 2x v_wmma_f32_16x16x32_f16
//   p = mask ? exp(e * 2/E) : 0   (unnormalized; logits are small -> safe)
//   P chunk (16x512) staged in LDS f16 (row stride 520 halves: conflict-free
//   ds_load_b128 for the A-operand re-read), O += P @ V via WMMA, normalize
//   by row sums (shfl_xor reduce within 16-lane halves) at the end.
// ---------------------------------------------------------------------------
__global__ void attn_kernel(const _Float16* __restrict__ qq,
                            const _Float16* __restrict__ kk,
                            const _Float16* __restrict__ vvT,
                            const int* __restrict__ mask,
                            _Float16* __restrict__ mid) {
  __shared__ _Float16 Pbuf[16 * 520];

  int b  = blockIdx.x;
  int qt = b & 127;
  int h  = (b >> 7) & 15;
  int n  = b >> 11;

  int lane   = threadIdx.x;
  int half16 = lane >> 4;   // 0: lanes 0-15, 1: lanes 16-31
  int l16    = lane & 15;

  size_t hb = ((size_t)(n * NH + h)) * SEQ;
  int q0 = qt * 16;
  const float scale = 2.0f / (float)EMB;   // reference: softmax(energy * 2/E)
  const int* mrow = mask + n * SEQ;

  // --- persistent Q A-fragments (K = d: frag0 d0..31, frag1 d32..63) ---
  v16h aq[2];
  {
    const _Float16* qrow = qq + (hb + q0 + l16) * HD;
#pragma unroll
    for (int c = 0; c < 2; ++c) {
      v8h lo = *(const v8h*)(qrow + c * 32 + half16 * 16);       // K c*32+{0..7|16..23}
      v8h hi = *(const v8h*)(qrow + c * 32 + half16 * 16 + 8);   // K c*32+{8..15|24..31}
#pragma unroll
      for (int i = 0; i < 8; ++i) { aq[c][i] = lo[i]; aq[c][8 + i] = hi[i]; }
    }
  }

  float rs[8];
#pragma unroll
  for (int r = 0; r < 8; ++r) rs[r] = 0.f;
  v8f oacc[4];
#pragma unroll
  for (int t = 0; t < 4; ++t) oacc[t] = v8f_zero();

  for (int chunk = 0; chunk < 4; ++chunk) {
    int cbase = chunk * 512;

    // ---- scores + exp for 512 keys -> LDS ----
    for (int kt = 0; kt < 32; ++kt) {
      int bcol = cbase + kt * 16 + l16;              // this lane's key column
      const _Float16* krow = kk + (hb + bcol) * HD;  // column of B = kk row
      v16h b0 = *(const v16h*)(krow + half16 * 16);        // K=d 0..15 | 16..31
      v16h b1 = *(const v16h*)(krow + 32 + half16 * 16);   // K=d 32..47 | 48..63
      v8f c = v8f_zero();
      c = __builtin_amdgcn_wmma_f32_16x16x32_f16(false, aq[0], false, b0,
                                                 (short)0, c, false, false);
      c = __builtin_amdgcn_wmma_f32_16x16x32_f16(false, aq[1], false, b1,
                                                 (short)0, c, false, false);
      int mv    = mrow[bcol];
      int pbase = kt * 16 + l16;
#pragma unroll
      for (int r = 0; r < 8; ++r) {
        float p = mv ? __expf(c[r] * scale) : 0.0f;  // masked: exp(-1e20*2/E)=0
        rs[r] += p;
        Pbuf[(r + half16 * 8) * 520 + pbase] = (_Float16)p;
      }
    }

    // ---- O += P(16x512) @ V(512x64) ----
    for (int kb = 0; kb < 512; kb += 32) {
      v16h a;
      {
        const _Float16* prow = Pbuf + l16 * 520 + kb + half16 * 16;
        v8h lo = *(const v8h*)(prow);        // K kb+{0..7 | 16..23}
        v8h hi = *(const v8h*)(prow + 8);    // K kb+{8..15 | 24..31}
#pragma unroll
        for (int i = 0; i < 8; ++i) { a[i] = lo[i]; a[8 + i] = hi[i]; }
      }
      int kglob = cbase + kb;
#pragma unroll
      for (int dt = 0; dt < 4; ++dt) {
        const _Float16* vcol = vvT +
            (((size_t)(n * NH + h)) * HD + dt * 16 + l16) * SEQ + kglob + half16 * 16;
        v16h bv = *(const v16h*)(vcol);      // column dt*16+l16, K rows kglob+0..31
        oacc[dt] = __builtin_amdgcn_wmma_f32_16x16x32_f16(false, a, false, bv,
                                                          (short)0, oacc[dt], false, false);
      }
    }
  }

  // ---- row-sum reduce within each 16-lane half, then normalize & store ----
#pragma unroll
  for (int r = 0; r < 8; ++r) {
    float s = rs[r];
    s += __shfl_xor(s, 1);
    s += __shfl_xor(s, 2);
    s += __shfl_xor(s, 4);
    s += __shfl_xor(s, 8);
    rs[r] = 1.0f / s;
  }
  int token0 = n * SEQ + q0;
#pragma unroll
  for (int dt = 0; dt < 4; ++dt) {
#pragma unroll
    for (int r = 0; r < 8; ++r) {
      int row = r + half16 * 8;
      float o = oacc[dt][r] * rs[r];
      mid[(size_t)(token0 + row) * EMB + h * HD + dt * 16 + l16] = (_Float16)o;
    }
  }
}

// ---------------------------------------------------------------------------
// Kernel 4: final = mid(4096x1024) @ Wo^T + bo. One wave per 16x16 out tile.
// ---------------------------------------------------------------------------
__global__ void outproj_kernel(const _Float16* __restrict__ mid,
                               const _Float16* __restrict__ WoH,
                               const float* __restrict__ bo,
                               float* __restrict__ out) {
  int b  = blockIdx.x;
  int et = b & 63;
  int tt = b >> 6;
  int lane   = threadIdx.x;
  int half16 = lane >> 4;
  int l16    = lane & 15;
  int t0 = tt * 16, e0 = et * 16;

  const _Float16* arow = mid + (size_t)(t0 + l16) * EMB;   // A row (token)
  const _Float16* brow = WoH + (size_t)(e0 + l16) * EMB;   // B col e0+l16 = Wo row

  v8f c = v8f_zero();
  for (int k = 0; k < EMB; k += 32) {
    v16h a;
    v8h lo = *(const v8h*)(arow + k + half16 * 16);
    v8h hi = *(const v8h*)(arow + k + half16 * 16 + 8);
#pragma unroll
    for (int i = 0; i < 8; ++i) { a[i] = lo[i]; a[8 + i] = hi[i]; }
    v16h bf = *(const v16h*)(brow + k + half16 * 16);      // K rows k+0..15 | k+16..31
    c = __builtin_amdgcn_wmma_f32_16x16x32_f16(false, a, false, bf,
                                               (short)0, c, false, false);
  }
  float bias = bo[e0 + l16];
#pragma unroll
  for (int r = 0; r < 8; ++r) {
    out[(size_t)(t0 + r + half16 * 8) * EMB + e0 + l16] = c[r] + bias;
  }
}

// ---------------------------------------------------------------------------
extern "C" void kernel_launch(void* const* d_in, const int* in_sizes, int n_in,
                              void* d_out, int out_size, void* d_ws, size_t ws_size,
                              hipStream_t stream) {
  (void)in_sizes; (void)n_in; (void)out_size; (void)ws_size;
  const float* key  = (const float*)d_in[0];
  // d_in[1] (query) and d_in[2] (value) are unused: reference overwrites both
  // with reshapes of `key`.
  const int*   mask = (const int*)d_in[3];
  const float* Wk   = (const float*)d_in[4];
  const float* Wq   = (const float*)d_in[5];
  const float* Wv   = (const float*)d_in[6];
  const float* Wo   = (const float*)d_in[7];
  const float* bo   = (const float*)d_in[8];
  float* out = (float*)d_out;

  char* ws = (char*)d_ws;
  const size_t perT = (size_t)NB * NH * SEQ * HD;      // 4,194,304 elements
  _Float16* qq  = (_Float16*)(ws);                      //  8 MB
  _Float16* kkp = (_Float16*)(ws + perT * 2);           //  8 MB
  _Float16* vvT = (_Float16*)(ws + perT * 4);           //  8 MB
  _Float16* mid = (_Float16*)(ws + perT * 6);           //  8 MB
  _Float16* WoH = (_Float16*)(ws + perT * 8);           //  2 MB  (total 34 MB)

  // 1) per-head projections
  proj_kernel<<<dim3((unsigned)(perT / 256)), dim3(256), 0, stream>>>(
      key, Wk, Wq, Wv, qq, kkp, vvT);
  // 2) Wo -> f16
  cvt_wo<<<dim3((EMB * EMB) / 256), dim3(256), 0, stream>>>(Wo, WoH);
  // 3) attention: N*H*(S/16) = 4096 waves
  attn_kernel<<<dim3(NB * NH * (SEQ / 16)), dim3(32), 0, stream>>>(
      qq, kkp, vvT, mask, mid);
  // 4) output projection: 256 token-tiles x 64 e-tiles
  outproj_kernel<<<dim3((NB * SEQ / 16) * (EMB / 16)), dim3(32), 0, stream>>>(
      mid, WoH, bo, out);
}
